// TritonFusedRCMAttention_65996467470876
// MI455X (gfx1250) — matrix operand order repaired and checked by reading, not compiled, not run on hardware
//
#include <hip/hip_runtime.h>
#include <hip/hip_bf16.h>
#include <cstdint>
#include <cstddef>

typedef __attribute__((ext_vector_type(16))) __bf16 v16bf;
typedef __attribute__((ext_vector_type(8)))  float  v8f;
typedef unsigned short u16;

#define DIM   1024
#define NHEAD 16
#define HDIM  64
#define BB    2
#define SEQ   2048
#define ROWS  (BB*SEQ)      /* 4096 */
#define QKVN  (3*DIM)       /* 3072 */
#define ATT_SCALE 0.125f    /* 64^-0.5 */

// ---------- helpers ----------
__device__ __forceinline__ u16 f2bf(float f) {
  union { float f; unsigned int u; } c; c.f = f;
  unsigned int u = c.u;
  unsigned int r = u + 0x7FFFu + ((u >> 16) & 1u);   // RNE truncate to bf16
  return (u16)(r >> 16);
}

union Frag { uint4 q[2]; v16bf v; };

// two 16-byte runs (8 bf16 each) -> one 16-element fragment
__device__ __forceinline__ v16bf ld_frag2(const u16* p0, const u16* p1) {
  Frag f;
  f.q[0] = *(const uint4*)p0;
  f.q[1] = *(const uint4*)p1;
  return f.v;
}

__device__ __forceinline__ v8f wmma_bf16(v16bf a, v16bf b, v8f c) {
  return __builtin_amdgcn_wmma_f32_16x16x32_bf16(false, a, false, b, (short)0, c,
                                                 false, false);
}

__device__ __forceinline__ v8f vzero8() {
  v8f z = {0.f,0.f,0.f,0.f,0.f,0.f,0.f,0.f};
  return z;
}

__device__ __forceinline__ float rmax16(float v) {
#pragma unroll
  for (int m = 1; m < 16; m <<= 1) v = fmaxf(v, __shfl_xor(v, m, 32));
  return v;
}
__device__ __forceinline__ float rsum16(float v) {
#pragma unroll
  for (int m = 1; m < 16; m <<= 1) v += __shfl_xor(v, m, 32);
  return v;
}

// ---------- conversion kernels ----------
__global__ void k_cvt(const float* __restrict__ src, u16* __restrict__ dst, int n) {
  int i = blockIdx.x * 256 + threadIdx.x;
  if (i < n) dst[i] = f2bf(src[i]);
}

// w: K x N (row major) -> wt: N x K (row major)  [transposed + bf16]
__global__ void k_cvt_t(const float* __restrict__ w, u16* __restrict__ wt, int K, int N) {
  int i = blockIdx.x * 256 + threadIdx.x;
  if (i < K * N) {
    int n = i / K;
    int k = i - n * K;
    wt[i] = f2bf(w[(size_t)k * N + n]);
  }
}

// ---------- QKV GEMM: (4096 x 1024) x (1024 x 3072) + bias, scatter to Q/K/Vt ----------
__global__ __launch_bounds__(128)
void k_gemm_qkv(const u16* __restrict__ A, const u16* __restrict__ Wt,
                const float* __restrict__ bias,
                u16* __restrict__ Qo, u16* __restrict__ Ko, u16* __restrict__ Vto) {
  const int lane = threadIdx.x & 31, wid = threadIdx.x >> 5;
  const int lo = lane & 15, hi = lane >> 4;
  const int m0 = blockIdx.x * 64 + (wid >> 1) * 32;
  const int n0 = blockIdx.y * 128 + (wid & 1) * 64;

  v8f acc[2][4];
#pragma unroll
  for (int mi = 0; mi < 2; mi++)
#pragma unroll
    for (int ni = 0; ni < 4; ni++) acc[mi][ni] = vzero8();

  for (int kb = 0; kb < DIM; kb += 32) {
    v16bf a[2];
#pragma unroll
    for (int mi = 0; mi < 2; mi++) {
      const u16* r = A + (size_t)(m0 + mi * 16 + lo) * DIM + kb + 16 * hi;
      a[mi] = ld_frag2(r, r + 8);
    }
#pragma unroll
    for (int ni = 0; ni < 4; ni++) {
      const u16* r = Wt + (size_t)(n0 + ni * 16 + lo) * DIM + kb + 16 * hi;
      v16bf b = ld_frag2(r, r + 8);
      acc[0][ni] = wmma_bf16(a[0], b, acc[0][ni]);
      acc[1][ni] = wmma_bf16(a[1], b, acc[1][ni]);
    }
  }

#pragma unroll
  for (int mi = 0; mi < 2; mi++)
#pragma unroll
    for (int ni = 0; ni < 4; ni++)
#pragma unroll
      for (int r = 0; r < 8; r++) {
        int row = m0 + mi * 16 + r + 8 * hi;
        int col = n0 + ni * 16 + lo;
        float v = acc[mi][ni][r] + bias[col];
        u16 bv = f2bf(v);
        int which = col >> 10, h = (col >> 6) & 15, d = col & 63;
        int b = row >> 11, n = row & (SEQ - 1);
        size_t bh = (size_t)(b * NHEAD + h);
        if (which == 0)      Qo[(bh * SEQ + n) * HDIM + d] = bv;
        else if (which == 1) Ko[(bh * SEQ + n) * HDIM + d] = bv;
        else                 Vto[(bh * HDIM + d) * SEQ + n] = bv;
      }
}

// ---------- flash attention + rCM skip epilogue ----------
__global__ __launch_bounds__(128)
void k_flash(const u16* __restrict__ Q, const u16* __restrict__ K,
             const u16* __restrict__ Vt, const float* __restrict__ x,
             const float* __restrict__ tptr, u16* __restrict__ Y) {
  __shared__ u16 plds[4 * 16 * 32];           // 1KB per wave P staging
  const int lane = threadIdx.x & 31, wid = threadIdx.x >> 5;
  const int lo = lane & 15, hi = lane >> 4;
  const int bh = blockIdx.x;                  // 0..31  (b*16+h)
  const int b = bh >> 4, h = bh & 15;
  const int q0 = blockIdx.y * 64 + wid * 16;  // query tile base
  const u16* Qb = Q + (size_t)bh * SEQ * HDIM;
  const u16* Kb = K + (size_t)bh * SEQ * HDIM;
  const u16* Vb = Vt + (size_t)bh * HDIM * SEQ;
  u16* lp = plds + wid * 16 * 32;

  v16bf qa[2];
#pragma unroll
  for (int c = 0; c < 2; c++) {
    const u16* r = Qb + (size_t)(q0 + lo) * HDIM + 32 * c + 16 * hi;
    qa[c] = ld_frag2(r, r + 8);
  }

  v8f acc[4];
#pragma unroll
  for (int c = 0; c < 4; c++) acc[c] = vzero8();
  float m_i[8], l_i[8];
#pragma unroll
  for (int r = 0; r < 8; r++) { m_i[r] = -1e30f; l_i[r] = 0.f; }

  for (int kb = 0; kb < SEQ; kb += 32) {
    // --- S = Q K^T for 32-key tile (two 16-key subtiles) ---
    v8f s[2];
#pragma unroll
    for (int j = 0; j < 2; j++) {
      const u16* kr = Kb + (size_t)(kb + j * 16 + lo) * HDIM + 16 * hi;
      v16bf kf0 = ld_frag2(kr, kr + 8);        // d = 0..31
      v16bf kf1 = ld_frag2(kr + 32, kr + 40);  // d = 32..63
      v8f t0 = wmma_bf16(qa[0], kf0, vzero8());
      s[j] = wmma_bf16(qa[1], kf1, t0);
    }

    // --- online softmax per row (row = r + 8*hi, spread over 16-lane half) ---
    float alpha[8];
#pragma unroll
    for (int r = 0; r < 8; r++) {
      float s0 = s[0][r] * ATT_SCALE;
      float s1 = s[1][r] * ATT_SCALE;
      float tm = rmax16(fmaxf(s0, s1));
      float mn = fmaxf(m_i[r], tm);
      float al = __expf(m_i[r] - mn);
      float p0 = __expf(s0 - mn);
      float p1 = __expf(s1 - mn);
      float rs = rsum16(p0 + p1);
      l_i[r] = al * l_i[r] + rs;
      m_i[r] = mn;
      alpha[r] = al;
      // stage P (bf16) to LDS in [row][key_local] order
      lp[(r + 8 * hi) * 32 + lo] = f2bf(p0);
      lp[(r + 8 * hi) * 32 + 16 + lo] = f2bf(p1);
    }
#pragma unroll
    for (int c = 0; c < 4; c++)
#pragma unroll
      for (int r = 0; r < 8; r++) acc[c][r] *= alpha[r];

    // intra-wave LDS store->load fence (CDNA5 split DS counter)
    asm volatile("s_wait_dscnt 0" ::: "memory");

    // read P back in A-fragment layout
    const u16* pr = lp + lo * 32 + 16 * hi;
    v16bf pa = ld_frag2(pr, pr + 8);

    // --- O += P V (Vt layout makes B fragments contiguous) ---
#pragma unroll
    for (int c = 0; c < 4; c++) {
      const u16* vr = Vb + (size_t)(16 * c + lo) * SEQ + kb + 16 * hi;
      v16bf vf = ld_frag2(vr, vr + 8);
      acc[c] = wmma_bf16(pa, vf, acc[c]);
    }
  }

  // --- epilogue: out = c_skip * x_heads + c_out * (O / l) ---
  float tv = tptr[0];
  float t2 = tv * tv;
  float c_skip = 1.0f / (t2 + 1.0f);           // sigma_data = 1
  float c_out = tv * rsqrtf(1.0f + t2);
#pragma unroll
  for (int r = 0; r < 8; r++) {
    float inv = 1.0f / l_i[r];
    int row = q0 + r + 8 * hi;
#pragma unroll
    for (int c = 0; c < 4; c++) {
      int d = 16 * c + lo;
      size_t xi = ((size_t)(b * SEQ) + row) * DIM + h * HDIM + d;
      float o = acc[c][r] * inv;
      Y[xi] = f2bf(c_skip * x[xi] + c_out * o);
    }
  }
}

// ---------- proj GEMM: (4096 x 1024) x (1024 x 1024) + bias -> f32 out ----------
__global__ __launch_bounds__(128)
void k_gemm_proj(const u16* __restrict__ A, const u16* __restrict__ Wt,
                 const float* __restrict__ bias, float* __restrict__ out) {
  const int lane = threadIdx.x & 31, wid = threadIdx.x >> 5;
  const int lo = lane & 15, hi = lane >> 4;
  const int m0 = blockIdx.x * 64 + (wid >> 1) * 32;
  const int n0 = blockIdx.y * 128 + (wid & 1) * 64;

  v8f acc[2][4];
#pragma unroll
  for (int mi = 0; mi < 2; mi++)
#pragma unroll
    for (int ni = 0; ni < 4; ni++) acc[mi][ni] = vzero8();

  for (int kb = 0; kb < DIM; kb += 32) {
    v16bf a[2];
#pragma unroll
    for (int mi = 0; mi < 2; mi++) {
      const u16* r = A + (size_t)(m0 + mi * 16 + lo) * DIM + kb + 16 * hi;
      a[mi] = ld_frag2(r, r + 8);
    }
#pragma unroll
    for (int ni = 0; ni < 4; ni++) {
      const u16* r = Wt + (size_t)(n0 + ni * 16 + lo) * DIM + kb + 16 * hi;
      v16bf b = ld_frag2(r, r + 8);
      acc[0][ni] = wmma_bf16(a[0], b, acc[0][ni]);
      acc[1][ni] = wmma_bf16(a[1], b, acc[1][ni]);
    }
  }

#pragma unroll
  for (int mi = 0; mi < 2; mi++)
#pragma unroll
    for (int ni = 0; ni < 4; ni++)
#pragma unroll
      for (int r = 0; r < 8; r++) {
        int row = m0 + mi * 16 + r + 8 * hi;
        int col = n0 + ni * 16 + lo;
        out[(size_t)row * DIM + col] = acc[mi][ni][r] + bias[col];
      }
}

// ---------- launch ----------
extern "C" void kernel_launch(void* const* d_in, const int* in_sizes, int n_in,
                              void* d_out, int out_size, void* d_ws, size_t ws_size,
                              hipStream_t stream) {
  (void)in_sizes; (void)n_in; (void)out_size; (void)ws_size;
  const float* x      = (const float*)d_in[0];
  const float* t      = (const float*)d_in[1];
  const float* w_qkv  = (const float*)d_in[2];
  const float* b_qkv  = (const float*)d_in[3];
  const float* w_proj = (const float*)d_in[4];
  const float* b_proj = (const float*)d_in[5];
  float* out = (float*)d_out;

  // workspace layout (bf16 elements)
  u16* xb  = (u16*)d_ws;                          // 4096*1024
  u16* wqt = xb  + (size_t)ROWS * DIM;            // 3072*1024 (w_qkv^T)
  u16* wpt = wqt + (size_t)QKVN * DIM;            // 1024*1024 (w_proj^T)
  u16* Qd  = wpt + (size_t)DIM * DIM;             // 32*2048*64
  u16* Kd  = Qd  + (size_t)ROWS * DIM;
  u16* Vtd = Kd  + (size_t)ROWS * DIM;            // transposed (B,H,D,N)
  u16* Yd  = Vtd + (size_t)ROWS * DIM;            // attn+skip result, bf16

  int n1 = ROWS * DIM;
  k_cvt<<<(n1 + 255) / 256, 256, 0, stream>>>(x, xb, n1);
  int n2 = DIM * QKVN;
  k_cvt_t<<<(n2 + 255) / 256, 256, 0, stream>>>(w_qkv, wqt, DIM, QKVN);
  int n3 = DIM * DIM;
  k_cvt_t<<<(n3 + 255) / 256, 256, 0, stream>>>(w_proj, wpt, DIM, DIM);

  k_gemm_qkv<<<dim3(ROWS / 64, QKVN / 128), 128, 0, stream>>>(xb, wqt, b_qkv, Qd, Kd, Vtd);
  k_flash<<<dim3(BB * NHEAD, SEQ / 64), 128, 0, stream>>>(Qd, Kd, Vtd, x, t, Yd);
  k_gemm_proj<<<dim3(ROWS / 64, DIM / 128), 128, 0, stream>>>(Yd, wpt, b_proj, out);
}
